// Patcher_9414568313317
// MI455X (gfx1250) — compile-verified, matrix operand order checked by reading, not run on hardware
//
#include <hip/hip_runtime.h>
#include <hip/hip_bf16.h>

// ---------------------------------------------------------------------------
// Patcher: rotated-patch overlay augmentation for MI455X (gfx1250).
// Memory-bound (~85 MB traffic, ~4us at 23.3 TB/s HBM; whole working set fits
// in 192MB L2). Single streaming pass over pixels; WMMA used for the two mean
// reductions (ones-matrix row-sum trick, f32 accumulate); box-parameter table
// staged into LDS via GLOBAL_LOAD_ASYNC_TO_LDS (builtin confirmed present).
// ---------------------------------------------------------------------------

#define BB 8
#define HH 640
#define WW 640
#define NN 12
#define PP 256
#define HW3 (HH * WW * 3)      // 1,228,800
#define PP3 (PP * PP * 3)      // 196,608
#define BPFLOATS (NN * 16)     // 192 floats of box params per image

typedef __attribute__((ext_vector_type(16))) _Float16 v16h;
typedef __attribute__((ext_vector_type(8)))  float    v8f;

#define AS1 __attribute__((address_space(1)))
#define AS3 __attribute__((address_space(3)))

// --- async copy to LDS: feature-detect builtins, never break the compile ---
#if __has_builtin(__builtin_amdgcn_global_load_async_to_lds_b32)
#define HAVE_ASYNC_LDS 1
#endif

__device__ __forceinline__ void wait_asynccnt0() {
#if __has_builtin(__builtin_amdgcn_s_wait_asynccnt)
    __builtin_amdgcn_s_wait_asynccnt(0);
#else
    asm volatile("s_wait_asynccnt 0" ::: "memory");
#endif
}

// -------------------------- threefry2x32 RNG -------------------------------
__device__ __forceinline__ unsigned rotl32(unsigned x, unsigned n) {
    return (x << n) | (x >> (32u - n));
}

__device__ __forceinline__ uint2 mku2(unsigned a, unsigned b) {
    uint2 r; r.x = a; r.y = b; return r;
}

__device__ inline uint2 tf2x32(uint2 k, uint2 c) {
    unsigned ks0 = k.x, ks1 = k.y, ks2 = 0x1BD11BDAu ^ ks0 ^ ks1;
    unsigned x0 = c.x + ks0, x1 = c.y + ks1;
#define TFR(r) { x0 += x1; x1 = rotl32(x1, r); x1 ^= x0; }
    TFR(13) TFR(15) TFR(26) TFR(6)
    x0 += ks1; x1 += ks2 + 1u;
    TFR(17) TFR(29) TFR(16) TFR(24)
    x0 += ks2; x1 += ks0 + 2u;
    TFR(13) TFR(15) TFR(26) TFR(6)
    x0 += ks0; x1 += ks1 + 3u;
    TFR(17) TFR(29) TFR(16) TFR(24)
    x0 += ks1; x1 += ks2 + 4u;
    TFR(13) TFR(15) TFR(26) TFR(6)
    x0 += ks2; x1 += ks0 + 5u;
#undef TFR
    return mku2(x0, x1);
}

__device__ __forceinline__ float u01(unsigned v) {
    return (float)(v >> 8) * 5.9604645e-8f;   // [0,1) from top 24 bits
}

__device__ __forceinline__ float clamp1(float x) {
    return fminf(fmaxf(x, -1.0f), 1.0f);
}

__device__ inline void boxmuller(float u1, float u2, float* n0, float* n1) {
    u1 = fmaxf(u1, 1e-9f);
    float r = sqrtf(-2.0f * logf(u1));
    float a = 6.28318530718f * u2;
    *n0 = r * __cosf(a);
    *n1 = r * __sinf(a);
}

// ---------------- WMMA ones-matrix sum reduction helper --------------------
// Each wave consumes 512 f32 elements per WMMA (lane loads 16 contiguous
// floats -> f16 A operand; B = ones). D[i][j] = sum_k A[i][k] + C[i][j],
// so one D column holds all row sums; summing column 0 across lanes 0 and 16
// gives the wave's total in f32 accumulation.

__device__ __forceinline__ v16h ones16h() {
    v16h o;
#pragma unroll
    for (int i = 0; i < 16; ++i) o[i] = (_Float16)1.0f;
    return o;
}

// ------------------------- kernel 1: image means ---------------------------
__global__ __launch_bounds__(512)
void mean_image_kernel(const float* __restrict__ img, float* __restrict__ meanImg) {
    const int b    = blockIdx.x;
    const int tid  = threadIdx.x;
    const int lane = tid & 31;
    const int wave = tid >> 5;
    const float* base = img + (size_t)b * HW3;

    const v16h ones = ones16h();
    v8f acc = {};
    // HW3 / (16 waves * 512) = 150 iterations
#pragma unroll 2
    for (int it = 0; it < 150; ++it) {
        size_t off = (size_t)it * 8192 + (size_t)wave * 512 + (size_t)lane * 16;
        const float4* f4 = (const float4*)(base + off);
        v16h a;
#pragma unroll
        for (int q = 0; q < 4; ++q) {
            float4 v = f4[q];
            a[4 * q + 0] = (_Float16)v.x;
            a[4 * q + 1] = (_Float16)v.y;
            a[4 * q + 2] = (_Float16)v.z;
            a[4 * q + 3] = (_Float16)v.w;
        }
        acc = __builtin_amdgcn_wmma_f32_16x16x32_f16(
            false, a, false, ones, (short)0, acc, false, false);
    }
    float s = acc[0] + acc[1] + acc[2] + acc[3] + acc[4] + acc[5] + acc[6] + acc[7];
    __shared__ float red[512];
    red[tid] = s;
    __syncthreads();
    if (tid == 0) {
        float t = 0.0f;
        for (int w = 0; w < 16; ++w) t += red[w * 32 + 0] + red[w * 32 + 16];
        meanImg[b] = t / (float)HW3;
    }
}

// ------------- kernel 2: per-image patch print-adjust + mean ---------------
__global__ __launch_bounds__(512)
void patch_adjust_kernel(const float* __restrict__ patch,
                         float* __restrict__ patchAdj,
                         float* __restrict__ meanPatch) {
    const int b    = blockIdx.x;
    const int tid  = threadIdx.x;
    const int lane = tid & 31;
    const int wave = tid >> 5;

    // per-image RNG: wmul = 0.5 + 0.1*N(0,1), badd = 0.01*N(0,1), 3 channels
    uint2 root = mku2(42u, 0u);
    uint2 imgk = tf2x32(root, mku2(1u, (unsigned)b));
    uint2 kw   = tf2x32(imgk, mku2(2u, 0u));
    uint2 kb   = tf2x32(imgk, mku2(2u, 1u));
    float wmul[3], badd[3];
    {
        uint2 rA = tf2x32(kw, mku2(0u, 0u));
        uint2 rB = tf2x32(kw, mku2(1u, 0u));
        float n0, n1, n2, nx;
        boxmuller(u01(rA.x), u01(rA.y), &n0, &n1);
        boxmuller(u01(rB.x), u01(rB.y), &n2, &nx);
        wmul[0] = 0.5f + 0.1f * n0;
        wmul[1] = 0.5f + 0.1f * n1;
        wmul[2] = 0.5f + 0.1f * n2;
        uint2 rC = tf2x32(kb, mku2(0u, 0u));
        uint2 rD = tf2x32(kb, mku2(1u, 0u));
        boxmuller(u01(rC.x), u01(rC.y), &n0, &n1);
        boxmuller(u01(rD.x), u01(rD.y), &n2, &nx);
        badd[0] = 0.01f * n0;
        badd[1] = 0.01f * n1;
        badd[2] = 0.01f * n2;
    }

    float* dstb = patchAdj + (size_t)b * PP3;
    const v16h ones = ones16h();
    v8f acc = {};
    // PP3 / 8192 = 24 iterations
    for (int it = 0; it < 24; ++it) {
        int off = it * 8192 + wave * 512 + lane * 16;
        v16h a;
        int ch = off % 3;
#pragma unroll
        for (int e = 0; e < 16; ++e) {
            float v  = patch[off + e];
            float pc = clamp1(wmul[ch] * v + badd[ch]);
            dstb[off + e] = pc;
            a[e] = (_Float16)pc;
            ch = (ch == 2) ? 0 : ch + 1;
        }
        acc = __builtin_amdgcn_wmma_f32_16x16x32_f16(
            false, a, false, ones, (short)0, acc, false, false);
    }
    float s = acc[0] + acc[1] + acc[2] + acc[3] + acc[4] + acc[5] + acc[6] + acc[7];
    __shared__ float red[512];
    red[tid] = s;
    __syncthreads();
    if (tid == 0) {
        float t = 0.0f;
        for (int w = 0; w < 16; ++w) t += red[w * 32 + 0] + red[w * 32 + 16];
        meanPatch[b] = t / (float)PP3;
    }
}

// --------------- kernel 3: brightness-match shift + clip -------------------
__global__ __launch_bounds__(256)
void patch_shift_kernel(float* __restrict__ patchAdj,
                        const float* __restrict__ meanImg,
                        const float* __restrict__ meanPatch) {
    int idx = blockIdx.x * 256 + threadIdx.x;   // grid sized exactly B*PP3
    int b = idx / PP3;
    float s = meanImg[b] - meanPatch[b];
    patchAdj[idx] = clamp1(patchAdj[idx] + s);
}

// --------------------- kernel 4: per-box parameters ------------------------
// record: [valid, y0f, x0f, diagf, c, ca, sa, top, phf, 256/safe_ph, delta,
//          kn.x bits, kn.y bits, pad, pad, pad]
__global__ __launch_bounds__(128)
void box_params_kernel(const float* __restrict__ boxes,
                       const float* __restrict__ scale_p,
                       float* __restrict__ bp) {
    int t = threadIdx.x;
    if (t >= BB * NN) return;
    int b = t / NN, n = t % NN;
    float scale = *scale_p;

    const float* bx = boxes + (size_t)(b * NN + n) * 4;
    float ymin = bx[0], xmin = bx[1], ymax = bx[2], xmax = bx[3];
    float h = ymax - ymin, w = xmax - xmin;
    float longer = fmaxf(h, w);
    float ps   = floorf(longer * scale);
    float diag = fminf(1.41421356237f * ps, (float)WW);

    uint2 root = mku2(42u, 0u);
    uint2 imgk = tf2x32(root, mku2(1u, (unsigned)b));
    uint2 ks   = tf2x32(imgk, mku2(2u, 2u));
    uint2 bk   = tf2x32(ks,   mku2(3u, (unsigned)n));
    uint2 kc1  = tf2x32(bk, mku2(4u, 0u));
    uint2 kc2  = tf2x32(bk, mku2(4u, 1u));
    uint2 ka   = tf2x32(bk, mku2(4u, 2u));
    uint2 kd   = tf2x32(bk, mku2(4u, 3u));
    uint2 kn   = tf2x32(bk, mku2(4u, 4u));

    float jy = (u01(kc1.x) - 0.5f) * 0.2f * h;
    float jx = (u01(kc2.x) - 0.5f) * 0.2f * w;
    float oy = ymin + 0.5f * h + jy;
    float ox = xmin + 0.5f * w + jx;
    float ymin_p = fmaxf(oy - 0.5f * diag, 0.0f);
    float xmin_p = fmaxf(ox - 0.5f * diag, 0.0f);
    if (ymin_p + diag > (float)HH) ymin_p = (float)HH - diag;
    if (xmin_p + diag > (float)WW) xmin_p = (float)WW - diag;

    float valid = (ps * ps > 4.0f) ? 1.0f : 0.0f;
    float y0f   = floorf(ymin_p);
    float x0f   = floorf(xmin_p);
    float phf   = ps;
    float diagf = floorf(diag);
    float top   = floorf((diagf - phf) * 0.5f);
    float angle = (u01(ka.x) * 2.0f - 1.0f) * 0.349065850399f;   // 20 deg
    float delta = (u01(kd.x) * 2.0f - 1.0f) * 0.3f;
    float c     = (diagf - 1.0f) * 0.5f;

    float* q = bp + (size_t)t * 16;
    q[0]  = valid;
    q[1]  = y0f;
    q[2]  = x0f;
    q[3]  = diagf;
    q[4]  = c;
    q[5]  = __cosf(angle);
    q[6]  = __sinf(angle);
    q[7]  = top;
    q[8]  = phf;
    q[9]  = 256.0f / fmaxf(phf, 1.0f);
    q[10] = delta;
    q[11] = __uint_as_float(kn.x);
    q[12] = __uint_as_float(kn.y);
    q[13] = 0.0f; q[14] = 0.0f; q[15] = 0.0f;
}

// ---------------------- kernel 5: main warp/overlay ------------------------
__device__ inline void bilin3(const float* __restrict__ p2,
                              float py, float px, float out[3]) {
    float y0 = floorf(py), x0 = floorf(px);
    float wy = py - y0, wx = px - x0;
    int y0i = (int)y0; y0i = (y0i < 0) ? 0 : ((y0i > PP - 1) ? PP - 1 : y0i);
    int x0i = (int)x0; x0i = (x0i < 0) ? 0 : ((x0i > PP - 1) ? PP - 1 : x0i);
    int y1i = (y0i + 1 > PP - 1) ? PP - 1 : y0i + 1;
    int x1i = (x0i + 1 > PP - 1) ? PP - 1 : x0i + 1;
    const float* r00 = p2 + (size_t)(y0i * PP + x0i) * 3;
    const float* r01 = p2 + (size_t)(y0i * PP + x1i) * 3;
    const float* r10 = p2 + (size_t)(y1i * PP + x0i) * 3;
    const float* r11 = p2 + (size_t)(y1i * PP + x1i) * 3;
#pragma unroll
    for (int ch = 0; ch < 3; ++ch) {
        float top = (1.0f - wx) * r00[ch] + wx * r01[ch];
        float bot = (1.0f - wx) * r10[ch] + wx * r11[ch];
        out[ch] = (1.0f - wy) * top + wy * bot;
    }
}

__global__ __launch_bounds__(256)
void apply_kernel(const float* __restrict__ img,
                  const float* __restrict__ patchAdj,
                  const float* __restrict__ bp,
                  float* __restrict__ out) {
    const int b   = blockIdx.y;
    const int pix = blockIdx.x * 256 + threadIdx.x;   // grid sized exactly H*W
    const int y   = pix / WW;
    const int x   = pix - y * WW;

    // Stage this image's 12x16-float box-parameter table into LDS once per
    // block. Uses the gfx1250 async copy engine (GLOBAL_LOAD_ASYNC_TO_LDS_B32
    // + s_wait_asynccnt) when available; falls back to a plain LDS fill.
    __shared__ __align__(16) float qs[BPFLOATS];
    {
        const float* bpb = bp + (size_t)b * BPFLOATS;
        const int t = threadIdx.x;
#if defined(HAVE_ASYNC_LDS)
        if (t < BPFLOATS) {
            __builtin_amdgcn_global_load_async_to_lds_b32(
                (AS1 int*)(bpb + t), (AS3 int*)&qs[t], 0, 0);
        }
        wait_asynccnt0();
#else
        if (t < BPFLOATS) qs[t] = bpb[t];
#endif
        __syncthreads();
    }

    const float* p2 = patchAdj + (size_t)b * PP3;
    // Pull the gather source toward the caches early (global_prefetch_b8).
    __builtin_prefetch(p2 + (size_t)(threadIdx.x) * 768, 0, 3);

    const size_t ib = (size_t)b * HW3 + (size_t)pix * 3;
    float r  = img[ib + 0];
    float g  = img[ib + 1];
    float bl = img[ib + 2];

    const float yy = (float)y, xx = (float)x;

    // Last box wins -> iterate backwards, first hit terminates.
    for (int n = NN - 1; n >= 0; --n) {
        const float* q = &qs[n << 4];
        if (q[0] == 0.0f) continue;
        float y0f = q[1], x0f = q[2], diagf = q[3];
        float u = yy - y0f, v = xx - x0f;
        if (!((u >= 0.0f) & (u < diagf) & (v >= 0.0f) & (v < diagf))) continue;
        float cc = q[4], ca = q[5], sa = q[6];
        float top = q[7], phf = q[8], sc = q[9], delta = q[10];
        float sv = ca * (v - cc) - sa * (u - cc) + cc;
        float su = sa * (v - cc) + ca * (u - cc) + cc;
        float ry = su - top, rx = sv - top;
        if (!((ry >= 0.0f) & (ry <= phf - 1.0f) &
              (rx >= 0.0f) & (rx <= phf - 1.0f))) continue;
        float py = (ry + 0.5f) * sc - 0.5f;
        float px = (rx + 0.5f) * sc - 0.5f;
        float smp[3];
        bilin3(p2, py, px, smp);
        uint2 kn = mku2(__float_as_uint(q[11]), __float_as_uint(q[12]));
        uint2 r1 = tf2x32(kn, mku2((unsigned)pix, 0u));
        uint2 r2 = tf2x32(kn, mku2((unsigned)pix, 1u));
        float n0 = u01(r1.x) * 0.02f - 0.01f;
        float n1 = u01(r1.y) * 0.02f - 0.01f;
        float n2 = u01(r2.x) * 0.02f - 0.01f;
        r  = clamp1(smp[0] + n0 + delta);
        g  = clamp1(smp[1] + n1 + delta);
        bl = clamp1(smp[2] + n2 + delta);
        break;
    }

    out[ib + 0] = r;
    out[ib + 1] = g;
    out[ib + 2] = bl;
}

// ------------------------------ launcher -----------------------------------
extern "C" void kernel_launch(void* const* d_in, const int* in_sizes, int n_in,
                              void* d_out, int out_size, void* d_ws, size_t ws_size,
                              hipStream_t stream) {
    const float* images = (const float*)d_in[0];   // [B,H,W,3]
    const float* boxes  = (const float*)d_in[1];   // [B,N,4]
    const float* patch  = (const float*)d_in[2];   // [P,P,3]
    const float* scale  = (const float*)d_in[3];   // [1]
    float* out = (float*)d_out;

    char* ws = (char*)d_ws;
    float* meanImg   = (float*)(ws);                 // 8 floats
    float* meanPatch = (float*)(ws + 256);           // 8 floats
    float* boxPar    = (float*)(ws + 512);           // 96 * 16 floats
    float* patchAdj  = (float*)(ws + 16384);         // 8 * 196608 floats

    mean_image_kernel  <<<BB, 512, 0, stream>>>(images, meanImg);
    patch_adjust_kernel<<<BB, 512, 0, stream>>>(patch, patchAdj, meanPatch);
    patch_shift_kernel <<<(BB * PP3) / 256, 256, 0, stream>>>(patchAdj, meanImg, meanPatch);
    box_params_kernel  <<<1, 128, 0, stream>>>(boxes, scale, boxPar);
    apply_kernel       <<<dim3((HH * WW) / 256, BB), 256, 0, stream>>>(
        images, patchAdj, boxPar, out);
}
